// LSTMDecoder_68702296866914
// MI455X (gfx1250) — matrix operand order repaired and checked by reading, not compiled, not run on hardware
//
#include <hip/hip_runtime.h>
#include <hip/hip_bf16.h>

// ---------------------------------------------------------------------------
// LSTM decoder, CDNA5 (gfx1250), wave32.
//   prep_kernel      : fold 16-wide input proj into W_comb; bf16-pack weights
//   ctx_gate_kernel  : ctx_gate = context @ W_ihc^T + bias   (WMMA bf16)
//   recurrent_kernel : 256-step LSTM, c in WMMA accumulators (WMMA bf16)
//   head_kernel      : LayerNorm + cmd/coord heads (wave32 reductions)
// ---------------------------------------------------------------------------

#define Bn   256
#define Sn   256
#define DM   512
#define PR   128
#define G4   2048
#define KAUG 160          // 128 (h) + 16 (x16) + 16 (zero pad)

typedef __attribute__((ext_vector_type(16))) __bf16 v16bf;
typedef __attribute__((ext_vector_type(8)))  __bf16 v8bf;
typedef __attribute__((ext_vector_type(8)))  float  v8f;
typedef __attribute__((ext_vector_type(4)))  unsigned int u32x4;
typedef __attribute__((ext_vector_type(8)))  int i32x8;
typedef __attribute__((ext_vector_type(4)))  int i32x4;

// workspace layout (bytes)
#define OFF_WAUG   0u                      // 2048*160 bf16  = 655360
#define OFF_WIHC   655360u                 // 2048*128 bf16  = 524288
#define OFF_WHRB   1179648u                // 128*512  bf16  = 131072
#define OFF_BIAS   1310720u                // 2048 f32       = 8192
#define OFF_CTXF   1318912u                // 256*2048 f32   = 2097152 (fragment layout)
#define OFF_HS     3416064u                // 256*256*128 f32= 33554432

// recurrent LDS layout (dynamic shared)
#define CTX_B   131072u    // 16*2048 f32, WMMA-fragment order
#define AAUG_B  5120u      // 16*KAUG bf16
#define HF_B    16384u     // 16*512 bf16
#define SMEM_B  (CTX_B + AAUG_B + HF_B)

__device__ __forceinline__ float sigmf(float x) { return 1.0f / (1.0f + __expf(-x)); }

// A-fragment (16x32 bf16, ISA 7.12.2): lane<16 -> M=lane, K = {0..7,16..23}+0;
// lane>=16 -> M=lane-16, K = {8..15,24..31}. Two contiguous 8-elem chunks.
__device__ __forceinline__ v16bf frag_a_bf(const __bf16* rowp, int k0, int lane) {
  const int off = (lane & 16) ? 8 : 0;
  v8bf lo = *(const v8bf*)(rowp + k0 + off);
  v8bf hi = *(const v8bf*)(rowp + k0 + 16 + off);
  v16bf r;
#pragma unroll
  for (int i = 0; i < 8; ++i) { r[i] = lo[i]; r[8 + i] = hi[i]; }
  return r;
}

__device__ __forceinline__ v16bf frag_a_f32(const float* rowp, int k0, int lane) {
  const int off = (lane & 16) ? 8 : 0;
  v16bf r;
#pragma unroll
  for (int i = 0; i < 8; ++i) r[i] = (__bf16)rowp[k0 + off + i];
#pragma unroll
  for (int i = 0; i < 8; ++i) r[8 + i] = (__bf16)rowp[k0 + 16 + off + i];
  return r;
}

// B-fragment (32x16 bf16): lane -> N = lane&15, K = k0 + {0..7,16..23} (lane<16)
// or {8..15,24..31} (lane>=16), from row (n0 + N) of a row-major [N][K] weight.
__device__ __forceinline__ v16bf frag_b_bf(const __bf16* base, int stride, int n0,
                                           int k0, int lane) {
  const int row = n0 + (lane & 15);
  const int col = k0 + ((lane & 16) ? 8 : 0);
  const __bf16* p = base + (size_t)row * stride + col;
  v8bf lo = *(const v8bf*)(p);
  v8bf hi = *(const v8bf*)(p + 16);
  v16bf r;
#pragma unroll
  for (int i = 0; i < 8; ++i) { r[i] = lo[i]; r[8 + i] = hi[i]; }
  return r;
}

// ---------------------------------------------------------------------------
// prep: W_comb = W_ih[:, :512] @ [W_cmd|W_coord|W_head]  (exact refactoring),
// bias_gate = b_ih + b_hh + W_ih[:, :512] @ (b_cmd+b_coord+b_head),
// bf16-pack W_aug = [W_hh | W_comb | 0], W_ihc = W_ih[:,512:640], W_hrb = W_hr.
// ---------------------------------------------------------------------------
__global__ void prep_kernel(const float* __restrict__ W_cmd, const float* __restrict__ b_cmd,
                            const float* __restrict__ W_coord, const float* __restrict__ b_coord,
                            const float* __restrict__ W_head, const float* __restrict__ b_head,
                            const float* __restrict__ W_ih, const float* __restrict__ W_hh,
                            const float* __restrict__ W_hr,
                            const float* __restrict__ b_ih, const float* __restrict__ b_hh,
                            __bf16* __restrict__ W_aug, __bf16* __restrict__ W_ihc,
                            __bf16* __restrict__ W_hrb, float* __restrict__ bias_gate) {
  const int tid = threadIdx.x;
  if (blockIdx.x == 8) {                       // bf16-pack W_hr (128x512)
    for (int i = tid; i < PR * DM; i += blockDim.x) W_hrb[i] = (__bf16)W_hr[i];
    return;
  }
  const int g = blockIdx.x * 256 + tid;        // 0..2047
  float wc[16];
#pragma unroll
  for (int k = 0; k < 16; ++k) wc[k] = 0.0f;
  float bacc = 0.0f;
  const float* wrow = W_ih + (size_t)g * 640;
  for (int d = 0; d < DM; ++d) {
    const float w = wrow[d];
    bacc += w * (b_cmd[d] + b_coord[d] + b_head[d]);
#pragma unroll
    for (int k = 0; k < 6; ++k) wc[k] += w * W_cmd[d * 6 + k];
#pragma unroll
    for (int k = 0; k < 8; ++k) wc[6 + k] += w * W_coord[d * 8 + k];
#pragma unroll
    for (int k = 0; k < 2; ++k) wc[14 + k] += w * W_head[d * 2 + k];
  }
  bias_gate[g] = b_ih[g] + b_hh[g] + bacc;
  __bf16* arow = W_aug + (size_t)g * KAUG;
  for (int k = 0; k < PR; ++k) arow[k] = (__bf16)W_hh[(size_t)g * PR + k];
#pragma unroll
  for (int k = 0; k < 16; ++k) arow[128 + k] = (__bf16)wc[k];
#pragma unroll
  for (int k = 0; k < 16; ++k) arow[144 + k] = (__bf16)0.0f;
  for (int k = 0; k < PR; ++k) W_ihc[(size_t)g * PR + k] = (__bf16)wrow[512 + k];
}

// ---------------------------------------------------------------------------
// ctx_gate (256x2048) = context (256x128) @ W_ihc^T + bias_gate.
// Output stored in WMMA D-fragment order: [tile][lane][r] (32B contiguous / lane)
// so the recurrent kernel's accumulator init is a single v8f LDS load.
// One 16x16 tile per wave; 8 waves / block; 2048 tiles total.
// ---------------------------------------------------------------------------
__global__ void ctx_gate_kernel(const float* __restrict__ context,
                                const __bf16* __restrict__ W_ihc,
                                const float* __restrict__ bias_gate,
                                float* __restrict__ ctx_frag) {
  const int lane = threadIdx.x & 31;
  const int wv   = threadIdx.x >> 5;
  const int tile = blockIdx.x * 8 + wv;        // 0..2047 ( = bt*128 + nt )
  const int bt = tile >> 7, nt = tile & 127;
  const int m0 = bt * 16, n0 = nt * 16;
  const float* arow = context + (size_t)(m0 + (lane & 15)) * PR;
  v8f acc = {0, 0, 0, 0, 0, 0, 0, 0};
#pragma unroll
  for (int kk = 0; kk < 4; ++kk) {
    const int k0 = kk * 32;
    v16bf a = frag_a_f32(arow, k0, lane);
    v16bf b = frag_b_bf(W_ihc, PR, n0, k0, lane);
    acc = __builtin_amdgcn_wmma_f32_16x16x32_bf16(false, a, false, b,
                                                  (short)0, acc, false, false);
  }
  const float bias = bias_gate[n0 + (lane & 15)];
#pragma unroll
  for (int r = 0; r < 8; ++r) acc[r] += bias;
  *(v8f*)(ctx_frag + ((size_t)tile * 32 + lane) * 8) = acc;
}

// ---------------------------------------------------------------------------
// Recurrent kernel: 16 WGs (16 batches each, fully independent -> no grid sync),
// 1024 threads = 32 waves. Wave w owns cell units [16w,16w+16): 4 gate tiles
// (i/f/g/o) per step, 5 K-steps each over K=160 augmented operand. Cell state c
// stays in the wave's v8f accumulator registers for all 256 steps.
// ctx_gate slice (128 KB) pulled into LDS once via the Tensor Data Mover.
// ---------------------------------------------------------------------------
__global__ void __launch_bounds__(1024)
recurrent_kernel(const float* __restrict__ x_std,
                 const float* __restrict__ ctx_frag,
                 const __bf16* __restrict__ W_aug,
                 const __bf16* __restrict__ W_hrb,
                 float* __restrict__ hs) {
  extern __shared__ char smem[];
  float*  ctxf = (float*)smem;                         // [128 tiles][32][8]
  __bf16* Aaug = (__bf16*)(smem + CTX_B);              // [16][KAUG]
  __bf16* HF   = (__bf16*)(smem + CTX_B + AAUG_B);     // [16][512]

  const int tid  = threadIdx.x;
  const int lane = tid & 31;
  const int wv   = tid >> 5;                           // 0..31
  const int b0   = blockIdx.x * 16;
  const int Mhi  = (lane & 16) ? 8 : 0;

  // init: zero A_aug (h=0, pad=0)
  for (int i = tid; i < 16 * KAUG; i += blockDim.x) Aaug[i] = (__bf16)0.0f;

  // pull this batch-tile's ctx_gate fragments (32768 dwords) into LDS
  const float* ctx_src = ctx_frag + (size_t)blockIdx.x * 128 * 32 * 8;
#if __has_builtin(__builtin_amdgcn_tensor_load_to_lds)
  if (tid < 32) {                                      // one TDM issue (wave 0)
    unsigned long long ga = (unsigned long long)ctx_src;
    u32x4 g0;
    g0[0] = 1u;                                        // count=1
    g0[1] = 0u;                                        // lds_addr = 0 (ctx first)
    g0[2] = (unsigned)(ga & 0xffffffffu);
    g0[3] = (unsigned)((ga >> 32) & 0x1ffffffu) | (2u << 30);   // type=2
    const unsigned nd = 32768u;                        // dwords
    i32x8 g1;
    g1[0] = (int)(2u << 16);                           // data_size = 4B
    g1[1] = (int)((nd & 0xffffu) << 16);               // tensor_dim0 lo
    g1[2] = (int)(((nd >> 16) & 0xffffu) | (1u << 16)); // dim0 hi | tensor_dim1=1
    g1[3] = (int)((nd & 0xffffu) << 16);               // tile_dim0
    g1[4] = 1;                                         // tile_dim1=1
    g1[5] = (int)nd;                                   // tensor_dim0_stride
    g1[6] = 0; g1[7] = 0;
    i32x4 z4 = {0, 0, 0, 0};
    i32x8 z8 = {0, 0, 0, 0, 0, 0, 0, 0};
    // 6-arg form (clang-23 / therock-10.0 headers): g0, g1, g2, g3, g4, cpol
    __builtin_amdgcn_tensor_load_to_lds(g0, g1, z4, z4, z8, 0);
    __builtin_amdgcn_s_wait_tensorcnt(0);
  }
#else
  for (int i = tid; i < 32768; i += blockDim.x) ctxf[i] = ctx_src[i];
#endif
  __syncthreads();

  v8f c_acc = {0, 0, 0, 0, 0, 0, 0, 0};                // persistent cell state

  for (int s = 0; s < Sn; ++s) {
    // ---- stage x16 into A_aug cols [128..144)
    if (tid < 256) {
      const int b = tid >> 4, k = tid & 15;
      Aaug[b * KAUG + 128 + k] =
          (__bf16)x_std[((size_t)(b0 + b) * Sn + s) * 16 + k];
    }
    __syncthreads();

    // ---- gates = [h|x16] @ W_aug^T + ctx  (shared A-frags across 4 gates)
    v16bf afr[5];
#pragma unroll
    for (int kk = 0; kk < 5; ++kk)
      afr[kk] = frag_a_bf(Aaug + (lane & 15) * KAUG, kk * 32, lane);

    v8f gacc[4];
#pragma unroll
    for (int gi = 0; gi < 4; ++gi) {
      const int nb = gi * 512 + wv * 16;               // gate unit base
      const int nt = nb >> 4;
      v8f acc = *(const v8f*)(ctxf + ((size_t)nt * 32 + lane) * 8);
#pragma unroll
      for (int kk = 0; kk < 5; ++kk) {
        v16bf b = frag_b_bf(W_aug, KAUG, nb, kk * 32, lane);
        acc = __builtin_amdgcn_wmma_f32_16x16x32_bf16(false, afr[kk], false, b,
                                                      (short)0, acc, false, false);
      }
      gacc[gi] = acc;
    }

    // ---- cell update, h_full -> LDS (bf16)
#pragma unroll
    for (int r = 0; r < 8; ++r) {
      const float iv = sigmf(gacc[0][r]);
      const float fv = sigmf(gacc[1][r]);
      const float gv = tanhf(gacc[2][r]);
      const float ov = sigmf(gacc[3][r]);
      const float c  = fv * c_acc[r] + iv * gv;
      c_acc[r] = c;
      HF[(r + Mhi) * DM + (wv * 16 + (lane & 15))] = (__bf16)(ov * tanhf(c));
    }
    __syncthreads();

    // ---- h = h_full @ W_hr^T : 8 output tiles (128 proj), waves 0..7
    if (wv < 8) {
      v8f hacc = {0, 0, 0, 0, 0, 0, 0, 0};
#pragma unroll 4
      for (int kk = 0; kk < 16; ++kk) {
        v16bf a = frag_a_bf(HF + (lane & 15) * DM, kk * 32, lane);
        v16bf b = frag_b_bf(W_hrb, DM, wv * 16, kk * 32, lane);
        hacc = __builtin_amdgcn_wmma_f32_16x16x32_bf16(false, a, false, b,
                                                       (short)0, hacc, false, false);
      }
      const int p = wv * 16 + (lane & 15);
#pragma unroll
      for (int r = 0; r < 8; ++r) {
        const int M = r + Mhi;
        const float hv = hacc[r];
        hs[((size_t)(b0 + M) * Sn + s) * PR + p] = hv;  // fp32 for LayerNorm
        Aaug[M * KAUG + p] = (__bf16)hv;                // next step's A operand
      }
    }
    __syncthreads();
  }
}

// ---------------------------------------------------------------------------
// Head: per row (b,s): LayerNorm(128) + cmd head (6x128) + coord head (8x134).
// One wave32 per row; lane holds 4 values; shfl_xor tree reductions.
// ---------------------------------------------------------------------------
__global__ void head_kernel(const float* __restrict__ hs,
                            const float* __restrict__ ln_g, const float* __restrict__ ln_b,
                            const float* __restrict__ W_oc, const float* __restrict__ b_oc,
                            const float* __restrict__ W_oo, const float* __restrict__ b_oo,
                            float* __restrict__ out_cmd, float* __restrict__ out_crd) {
  const int lane = threadIdx.x & 31;
  const int row  = blockIdx.x * 8 + (threadIdx.x >> 5);   // 0..65535
  const float4 v4 = *(const float4*)(hs + (size_t)row * PR + lane * 4);
  const float v[4] = {v4.x, v4.y, v4.z, v4.w};
  float s1 = v[0] + v[1] + v[2] + v[3];
  float s2 = v[0] * v[0] + v[1] * v[1] + v[2] * v[2] + v[3] * v[3];
#pragma unroll
  for (int off = 16; off; off >>= 1) {
    s1 += __shfl_xor(s1, off);
    s2 += __shfl_xor(s2, off);
  }
  const float mean = s1 * (1.0f / 128.0f);
  const float rstd = rsqrtf(s2 * (1.0f / 128.0f) - mean * mean + 1e-5f);
  float hn[4];
#pragma unroll
  for (int j = 0; j < 4; ++j)
    hn[j] = (v[j] - mean) * rstd * ln_g[4 * lane + j] + ln_b[4 * lane + j];

  float cmd[6];
#pragma unroll
  for (int c = 0; c < 6; ++c) {
    float p = 0.0f;
#pragma unroll
    for (int j = 0; j < 4; ++j) p += hn[j] * W_oc[c * PR + 4 * lane + j];
#pragma unroll
    for (int off = 16; off; off >>= 1) p += __shfl_xor(p, off);
    cmd[c] = p + b_oc[c];
  }
  float crd[8];
#pragma unroll
  for (int c = 0; c < 8; ++c) {
    float p = 0.0f;
#pragma unroll
    for (int j = 0; j < 4; ++j) p += hn[j] * W_oo[c * 134 + 4 * lane + j];
#pragma unroll
    for (int off = 16; off; off >>= 1) p += __shfl_xor(p, off);
#pragma unroll
    for (int cc = 0; cc < 6; ++cc) p += cmd[cc] * W_oo[c * 134 + 128 + cc];
    crd[c] = p + b_oo[c];
  }
  if (lane == 0) {
#pragma unroll
    for (int c = 0; c < 6; ++c) out_cmd[(size_t)row * 6 + c] = cmd[c];
#pragma unroll
    for (int c = 0; c < 8; ++c) out_crd[(size_t)row * 8 + c] = crd[c];
  }
}

extern "C" void kernel_launch(void* const* d_in, const int* in_sizes, int n_in,
                              void* d_out, int out_size, void* d_ws, size_t ws_size,
                              hipStream_t stream) {
  const float* x_std   = (const float*)d_in[0];
  const float* context = (const float*)d_in[1];
  const float* W_cmd   = (const float*)d_in[2];
  const float* b_cmd   = (const float*)d_in[3];
  const float* W_coord = (const float*)d_in[4];
  const float* b_coord = (const float*)d_in[5];
  const float* W_head  = (const float*)d_in[6];
  const float* b_head  = (const float*)d_in[7];
  const float* W_ih    = (const float*)d_in[8];
  const float* W_hh    = (const float*)d_in[9];
  const float* W_hr    = (const float*)d_in[10];
  const float* b_ih    = (const float*)d_in[11];
  const float* b_hh    = (const float*)d_in[12];
  const float* ln_g    = (const float*)d_in[13];
  const float* ln_b    = (const float*)d_in[14];
  const float* W_oc    = (const float*)d_in[15];
  const float* b_oc    = (const float*)d_in[16];
  const float* W_oo    = (const float*)d_in[17];
  const float* b_oo    = (const float*)d_in[18];

  char* ws = (char*)d_ws;
  __bf16* W_aug   = (__bf16*)(ws + OFF_WAUG);
  __bf16* W_ihc   = (__bf16*)(ws + OFF_WIHC);
  __bf16* W_hrb   = (__bf16*)(ws + OFF_WHRB);
  float*  bias_g  = (float*)(ws + OFF_BIAS);
  float*  ctxfrag = (float*)(ws + OFF_CTXF);
  float*  hs      = (float*)(ws + OFF_HS);

  float* out_cmd = (float*)d_out;
  float* out_crd = (float*)d_out + (size_t)Bn * Sn * 6;

  prep_kernel<<<9, 256, 0, stream>>>(W_cmd, b_cmd, W_coord, b_coord, W_head, b_head,
                                     W_ih, W_hh, W_hr, b_ih, b_hh,
                                     W_aug, W_ihc, W_hrb, bias_g);
  ctx_gate_kernel<<<256, 256, 0, stream>>>(context, W_ihc, bias_g, ctxfrag);
  recurrent_kernel<<<16, 1024, SMEM_B, stream>>>(x_std, ctxfrag, W_aug, W_hrb, hs);
  head_kernel<<<8192, 256, 0, stream>>>(hs, ln_g, ln_b, W_oc, b_oc, W_oo, b_oo,
                                        out_cmd, out_crd);
}
